// New_LinkNet_2671469658457
// MI455X (gfx1250) — compile-verified, matrix operand order checked by reading, not run on hardware
//
#include <hip/hip_runtime.h>
#include <hip/hip_bf16.h>
#include <stdint.h>

#define N_NODES 20000
#define N_EDGES 160000
#define N_E2    320000

typedef __bf16 bf16_t;
typedef bf16_t v16bf __attribute__((ext_vector_type(16)));
typedef bf16_t v8bf  __attribute__((ext_vector_type(8)));
typedef float  v8f   __attribute__((ext_vector_type(8)));

__device__ __forceinline__ uint16_t f2bf(float f) {
  union { float f; uint32_t u; } c; c.f = f;
  uint32_t u = c.u;
  return (uint16_t)((u + 0x7FFFu + ((u >> 16) & 1u)) >> 16);   // RNE float->bf16
}
__device__ __forceinline__ float sigmoidf_(float x) { return 1.f / (1.f + expf(-x)); }
__device__ __forceinline__ void atomicMaxF(float* addr, float v) {
  if (v >= 0.f) atomicMax((int*)addr, __float_as_int(v));
  else          atomicMin((unsigned int*)addr, __float_as_uint(v));
}
// src(e) == ei[e]; dst(e) below (recv = concat(ei1, ei0))
__device__ __forceinline__ int dst_of(const int* ei, int e) {
  return (e < N_EDGES) ? ei[e + N_EDGES] : ei[e - N_EDGES];
}
// A-fragment (16x32 bf16, ISA layout): two contiguous 8-elem chunks per lane.
__device__ __forceinline__ v16bf load_a_frag(const uint16_t* rowp, int kb) {
  union { v16bf v; v8bf h[2]; } af;
  af.h[0] = *(const v8bf*)(rowp + kb);
  af.h[1] = *(const v8bf*)(rowp + kb + 16);
  return af.v;
}

// ---------------------------------------------------------------------------
// Weight pre-pack: fp32 row-major W[K,N] -> bf16 WMMA B-fragment tiles
// layout: [kt][nt][lane 0..31][elem 0..15]
// ---------------------------------------------------------------------------
__global__ void pack_w_kernel(const float* __restrict__ w, uint16_t* __restrict__ dst,
                              int K, int N, float scale) {
  int idx = blockIdx.x * blockDim.x + threadIdx.x;
  if (idx >= K * N) return;
  int i    = idx & 15;
  int lane = (idx >> 4) & 31;
  int tile = idx >> 9;
  int ntiles = N >> 4;
  int kt = tile / ntiles, nt = tile - kt * ntiles;
  int n = nt * 16 + (lane & 15);
  int k = kt * 32 + ((lane >> 4) << 3) + i + ((i >= 8) ? 8 : 0);
  dst[idx] = f2bf(w[(size_t)k * N + n] * scale);
}

__global__ void convert_bf_kernel(const float* __restrict__ src, uint16_t* __restrict__ dst,
                                  int n) {
  int i = blockIdx.x * blockDim.x + threadIdx.x;
  if (i < n) dst[i] = f2bf(src[i]);
}

// ---------------------------------------------------------------------------
// Node / edge embeddings (tiny K, scalar)
// ---------------------------------------------------------------------------
__global__ void node_embed_kernel(const float* __restrict__ x, const float* __restrict__ w,
                                  const float* __restrict__ b, const float* __restrict__ zemb,
                                  float* __restrict__ nf) {
  int idx = blockIdx.x * blockDim.x + threadIdx.x;
  if (idx >= N_NODES * 64) return;
  int node = idx >> 6, c = idx & 63;
  const float* xp = x + node * 6;
  float a = b[c];
#pragma unroll
  for (int k = 0; k < 6; ++k) a += xp[k] * w[k * 64 + c];
  a = fmaxf(a, 0.f);
  float zr = rintf(xp[2]);
  const float allowed[6] = {8.f, 23.f, 39.f, 54.f, 90.f, 180.f};
  int zi = 0; float best = fabsf(zr - allowed[0]);
#pragma unroll
  for (int t = 1; t < 6; ++t) { float d = fabsf(zr - allowed[t]); if (d < best) { best = d; zi = t; } }
  nf[idx] = a + zemb[zi * 64 + c];
}

__global__ void edge_embed_kernel(const float* __restrict__ x, const float* __restrict__ ea,
                                  const int* __restrict__ ei, const float* __restrict__ w,
                                  const float* __restrict__ b, float* __restrict__ ef) {
  int idx = blockIdx.x * blockDim.x + threadIdx.x;
  if (idx >= N_E2 * 64) return;
  int e = idx >> 6, c = idx & 63;
  int s = ei[e];
  int d = dst_of(ei, e);
  int eb = (e < N_EDGES) ? e : e - N_EDGES;
  float f[7];
#pragma unroll
  for (int j = 0; j < 4; ++j) f[j] = ea[(size_t)eb * 4 + j];
#pragma unroll
  for (int j = 0; j < 3; ++j) f[4 + j] = x[(size_t)s * 6 + j] - x[(size_t)d * 6 + j];
  float a = b[c];
#pragma unroll
  for (int j = 0; j < 7; ++j) a += f[j] * w[j * 64 + c];
  ef[idx] = fmaxf(a, 0.f);
}

// ---------------------------------------------------------------------------
// CBAM (channel + spatial attention), D = 64
// ---------------------------------------------------------------------------
__global__ void cbam_colreduce_kernel(const float* __restrict__ X, float* __restrict__ partial,
                                      int nrows) {
  int c = threadIdx.x;                 // 64 threads
  int r0 = blockIdx.x * 512;
  int r1 = r0 + 512; if (r1 > nrows) r1 = nrows;
  float m = -3.402823466e38f, s = 0.f;
  for (int r = r0; r < r1; ++r) { float v = X[(size_t)r * 64 + c]; m = fmaxf(m, v); s += v; }
  partial[(size_t)blockIdx.x * 128 + c]      = m;
  partial[(size_t)blockIdx.x * 128 + 64 + c] = s;
}

__global__ void cbam_stage2_kernel(const float* __restrict__ partial, int nblk,
                                   const float* __restrict__ w1, const float* __restrict__ b1,
                                   const float* __restrict__ w2, const float* __restrict__ b2,
                                   int nrows, float* __restrict__ ch) {
  __shared__ float pool[128];
  __shared__ float hid[4];
  int c = threadIdx.x;                 // 64 threads
  float m = -3.402823466e38f, s = 0.f;
  for (int b = 0; b < nblk; ++b) {
    m = fmaxf(m, partial[(size_t)b * 128 + c]);
    s += partial[(size_t)b * 128 + 64 + c];
  }
  pool[c]      = m;
  pool[64 + c] = s / (float)nrows;
  __syncthreads();
  if (c < 4) {
    float h = b1[c];
    for (int j = 0; j < 128; ++j) h += pool[j] * w1[j * 4 + c];
    hid[c] = fmaxf(h, 0.f);
  }
  __syncthreads();
  float o = b2[c];
#pragma unroll
  for (int h = 0; h < 4; ++h) o += hid[h] * w2[h * 64 + c];
  ch[c] = sigmoidf_(o);
}

__global__ void cbam_rowscale_kernel(float* __restrict__ X, const float* __restrict__ ch,
                                     float* __restrict__ rowstats, int nrows) {
  int r = blockIdx.x * blockDim.x + threadIdx.x;
  if (r >= nrows) return;
  float* p = X + (size_t)r * 64;
  float m = -3.402823466e38f, s = 0.f;
#pragma unroll 4
  for (int c = 0; c < 64; ++c) { float v = p[c] * ch[c]; p[c] = v; m = fmaxf(m, v); s += v; }
  rowstats[(size_t)r * 2]     = m;
  rowstats[(size_t)r * 2 + 1] = s * (1.f / 64.f);
}

__global__ void cbam_spatial_kernel(float* __restrict__ X, const float* __restrict__ rowstats,
                                    const float* __restrict__ cw, const float* __restrict__ cb,
                                    int nrows) {
  int r = blockIdx.x * blockDim.x + threadIdx.x;
  if (r >= nrows) return;
  float acc = cb[0];
#pragma unroll
  for (int t = 0; t < 7; ++t) {
    int rr = r + t - 3;
    if (rr < 0 || rr >= nrows) continue;
    acc += cw[t] * rowstats[(size_t)rr * 2] + cw[7 + t] * rowstats[(size_t)rr * 2 + 1];
  }
  float sg = sigmoidf_(acc);
  float* p = X + (size_t)r * 64;
#pragma unroll 4
  for (int c = 0; c < 64; ++c) p[c] *= sg;
}

// ---------------------------------------------------------------------------
// Fills / scatters / layernorms
// ---------------------------------------------------------------------------
__global__ void fill_kernel(float* __restrict__ p, float v, int n) {
  int i = blockIdx.x * blockDim.x + threadIdx.x;
  if (i < n) p[i] = v;
}
__global__ void fill_from_ptr_kernel(float* __restrict__ p, const float* __restrict__ v, int n) {
  int i = blockIdx.x * blockDim.x + threadIdx.x;
  if (i < n) p[i] = v[0];
}
__global__ void scatter_agg_kernel(const float* __restrict__ ef, const int* __restrict__ ei,
                                   float* __restrict__ agg) {
  int idx = blockIdx.x * blockDim.x + threadIdx.x;
  if (idx >= N_E2 * 64) return;
  int e = idx >> 6, c = idx & 63;
  atomicAdd(&agg[(size_t)dst_of(ei, e) * 64 + c], ef[idx]);
}

// LN(concat(nf, agg-nf)) -> bf16 comb (GEMM A operand only)
__global__ __launch_bounds__(256) void comb_ln_kernel(const float* __restrict__ nf,
                                                      const float* __restrict__ agg,
                                                      const float* __restrict__ g,
                                                      const float* __restrict__ b,
                                                      uint16_t* __restrict__ comb_bf) {
  int row = blockIdx.x * 8 + (threadIdx.x >> 5);
  if (row >= N_NODES) return;
  int lane = threadIdx.x & 31;
  float vals[4]; float s = 0.f, s2 = 0.f;
#pragma unroll
  for (int i = 0; i < 4; ++i) {
    int j = lane * 4 + i;
    float v;
    if (j < 64) v = nf[(size_t)row * 64 + j];
    else        v = agg[(size_t)row * 64 + (j - 64)] - nf[(size_t)row * 64 + (j - 64)];
    vals[i] = v; s += v; s2 += v * v;
  }
#pragma unroll
  for (int m = 1; m < 32; m <<= 1) { s += __shfl_xor(s, m, 32); s2 += __shfl_xor(s2, m, 32); }
  float mean = s * (1.f / 128.f);
  float var  = s2 * (1.f / 128.f) - mean * mean;
  float r    = rsqrtf(var + 1e-5f);
#pragma unroll
  for (int i = 0; i < 4; ++i) {
    int j = lane * 4 + i;
    comb_bf[(size_t)row * 128 + j] = f2bf((vals[i] - mean) * r * g[j] + b[j]);
  }
}

// LN over 256 of fp32 attention output -> bf16 out (GEMM A operand only)
__global__ __launch_bounds__(256) void ln256_kernel(const float* __restrict__ X,
                                                    const float* __restrict__ g,
                                                    const float* __restrict__ b,
                                                    uint16_t* __restrict__ out_bf) {
  int row = blockIdx.x * 8 + (threadIdx.x >> 5);
  if (row >= N_NODES) return;
  int lane = threadIdx.x & 31;
  const float* p = X + (size_t)row * 256 + lane * 8;
  float vals[8]; float s = 0.f, s2 = 0.f;
#pragma unroll
  for (int i = 0; i < 8; ++i) { float v = p[i]; vals[i] = v; s += v; s2 += v * v; }
#pragma unroll
  for (int m = 1; m < 32; m <<= 1) { s += __shfl_xor(s, m, 32); s2 += __shfl_xor(s2, m, 32); }
  float mean = s * (1.f / 256.f);
  float var  = s2 * (1.f / 256.f) - mean * mean;
  float r    = rsqrtf(var + 1e-5f);
  const float* gg = g + lane * 8; const float* bb = b + lane * 8;
  uint16_t* op = out_bf + (size_t)row * 256 + lane * 8;
#pragma unroll
  for (int i = 0; i < 8; ++i) op[i] = f2bf((vals[i] - mean) * r * gg[i] + bb[i]);
}

// ---------------------------------------------------------------------------
// Attention (softmax over incoming edges, 4 heads x 64)
// ---------------------------------------------------------------------------
__global__ void attn_alpha_kernel(const float* __restrict__ q, const float* __restrict__ k,
                                  const int* __restrict__ ei, float* __restrict__ alp,
                                  float* __restrict__ amax) {
  int idx = blockIdx.x * blockDim.x + threadIdx.x;
  if (idx >= N_E2 * 4) return;
  int e = idx >> 2, h = idx & 3;
  int s = ei[e], d = dst_of(ei, e);
  const float4* qp = (const float4*)(q + (size_t)d * 256 + h * 64);
  const float4* kp = (const float4*)(k + (size_t)s * 256 + h * 64);
  float a = 0.f;
#pragma unroll
  for (int t = 0; t < 16; ++t) {
    float4 qq = qp[t], kk = kp[t];
    a += qq.x * kk.x + qq.y * kk.y + qq.z * kk.z + qq.w * kk.w;
  }
  a *= 0.125f;                       // / sqrt(64)
  alp[idx] = a;
  atomicMaxF(&amax[d * 4 + h], a);
}

__global__ void attn_expsum_kernel(float* __restrict__ alp, const float* __restrict__ amax,
                                   const int* __restrict__ ei, float* __restrict__ asum) {
  int idx = blockIdx.x * blockDim.x + threadIdx.x;
  if (idx >= N_E2 * 4) return;
  int e = idx >> 2, h = idx & 3;
  int d = dst_of(ei, e);
  float a = expf(alp[idx] - amax[d * 4 + h]);
  alp[idx] = a;
  atomicAdd(&asum[d * 4 + h], a);
}

__global__ void attn_scatter_kernel(const float* __restrict__ alp, const float* __restrict__ asum,
                                    const float* __restrict__ v, const int* __restrict__ ei,
                                    float* __restrict__ outb) {
  int idx = blockIdx.x * blockDim.x + threadIdx.x;
  if (idx >= N_E2 * 256) return;
  int e = idx >> 8, c = idx & 255, h = c >> 6;
  int s = ei[e], d = dst_of(ei, e);
  float coeff = alp[e * 4 + h] / asum[d * 4 + h];
  atomicAdd(&outb[(size_t)d * 256 + c], coeff * v[(size_t)s * 256 + c]);
}

// ---------------------------------------------------------------------------
// WMMA GEMMs: one wave computes a 16x64 D slab (4 accumulators), bf16 A from
// pre-converted activations (two b128 loads per fragment, no cvt VALU).
// ---------------------------------------------------------------------------
__global__ __launch_bounds__(256) void gemm_qkvs_kernel(
    const uint16_t* __restrict__ comb_bf, const uint16_t* __restrict__ pk,
    const float* __restrict__ bq, const float* __restrict__ bk,
    const float* __restrict__ bv, const float* __restrict__ bs,
    float* __restrict__ q, float* __restrict__ k2,
    float* __restrict__ v2, float* __restrict__ s2) {
  int mat = blockIdx.y;
  int wid = blockIdx.x * 8 + (threadIdx.x >> 5);
  if (wid >= 1250 * 4) return;
  int mt = wid >> 2, ng = wid & 3;       // ng: group of 4 n-tiles
  int lane = threadIdx.x & 31;
  int m0 = mt * 16, koff = (lane >> 4) << 3;
  const uint16_t* ap = comb_bf + (size_t)(m0 + (lane & 15)) * 128;
  const uint16_t* wp = pk + (size_t)mat * (128 * 256);
  v8f acc[4] = {{}, {}, {}, {}};
#pragma unroll
  for (int kt = 0; kt < 4; ++kt) {
    v16bf af = load_a_frag(ap, kt * 32 + koff);
#pragma unroll
    for (int t = 0; t < 4; ++t) {
      int nt = ng * 4 + t;
      v16bf bfrag = *(const v16bf*)(wp + (((size_t)kt * 16 + nt) * 32 + lane) * 16);
      acc[t] = __builtin_amdgcn_wmma_f32_16x16x32_bf16(false, af, false, bfrag, (short)0,
                                                       acc[t], false, false);
    }
  }
  const float* bias4[4] = {bq, bk, bv, bs};
  float* out4[4] = {q, k2, v2, s2};
  int rbase = m0 + koff;
  float* op = out4[mat];
  const float* bp = bias4[mat];
#pragma unroll
  for (int t = 0; t < 4; ++t) {
    int col = (ng * 4 + t) * 16 + (lane & 15);
    float bias = bp[col];
#pragma unroll
    for (int j = 0; j < 8; ++j)
      op[(size_t)(rbase + j) * 256 + col] = acc[t][j] + bias;
  }
}

__global__ __launch_bounds__(256) void gemm_pn_kernel(
    const uint16_t* __restrict__ out_bf, const uint16_t* __restrict__ wp,
    const float* __restrict__ bias, float* __restrict__ pn,
    uint16_t* __restrict__ pn_bf) {
  int wid = blockIdx.x * 8 + (threadIdx.x >> 5);   // wave = full 16x64 slab
  if (wid >= 1250) return;
  int lane = threadIdx.x & 31;
  int m0 = wid * 16, koff = (lane >> 4) << 3;
  const uint16_t* ap = out_bf + (size_t)(m0 + (lane & 15)) * 256;
  v8f acc[4] = {{}, {}, {}, {}};
#pragma unroll
  for (int kt = 0; kt < 8; ++kt) {
    v16bf af = load_a_frag(ap, kt * 32 + koff);
#pragma unroll
    for (int t = 0; t < 4; ++t) {
      v16bf bfrag = *(const v16bf*)(wp + (((size_t)kt * 4 + t) * 32 + lane) * 16);
      acc[t] = __builtin_amdgcn_wmma_f32_16x16x32_bf16(false, af, false, bfrag, (short)0,
                                                       acc[t], false, false);
    }
  }
  int rbase = m0 + koff;
#pragma unroll
  for (int t = 0; t < 4; ++t) {
    int col = t * 16 + (lane & 15);
    float b = bias[col];
#pragma unroll
    for (int j = 0; j < 8; ++j) {
      float v = acc[t][j] + b;
      size_t o = (size_t)(rbase + j) * 64 + col;
      pn[o]    = v;
      pn_bf[o] = f2bf(v);
    }
  }
}

__global__ __launch_bounds__(256) void gemm_gate_kernel(
    const float* __restrict__ nf, const float* __restrict__ pn,
    const uint16_t* __restrict__ nf_bf, const uint16_t* __restrict__ pn_bf,
    const uint16_t* __restrict__ wp, const float* __restrict__ bias,
    float* __restrict__ nf_out, uint16_t* __restrict__ nf_out_bf) {
  int wid = blockIdx.x * 8 + (threadIdx.x >> 5);
  if (wid >= 1250) return;
  int lane = threadIdx.x & 31;
  int m0 = wid * 16, koff = (lane >> 4) << 3;
  int arow = m0 + (lane & 15);
  const uint16_t* a0 = nf_bf + (size_t)arow * 64;
  const uint16_t* a1 = pn_bf + (size_t)arow * 64;
  v8f acc[4] = {{}, {}, {}, {}};
#pragma unroll
  for (int kt = 0; kt < 4; ++kt) {
    // kt 0..1: k in [0,64) -> nf ; kt 2..3: k-64 -> pn   (chunks never straddle)
    v16bf af = (kt < 2) ? load_a_frag(a0, kt * 32 + koff)
                        : load_a_frag(a1, (kt - 2) * 32 + koff);
#pragma unroll
    for (int t = 0; t < 4; ++t) {
      v16bf bfrag = *(const v16bf*)(wp + (((size_t)kt * 4 + t) * 32 + lane) * 16);
      acc[t] = __builtin_amdgcn_wmma_f32_16x16x32_bf16(false, af, false, bfrag, (short)0,
                                                       acc[t], false, false);
    }
  }
  int rbase = m0 + koff;
#pragma unroll
  for (int t = 0; t < 4; ++t) {
    int col = t * 16 + (lane & 15);
    float b = bias[col];
#pragma unroll
    for (int j = 0; j < 8; ++j) {
      int r = rbase + j;
      float g  = sigmoidf_(acc[t][j] + b);
      float nv = nf[(size_t)r * 64 + col];
      float pv = pn[(size_t)r * 64 + col];
      // nf_new = nf*g + pn*(1-g) + x0, x0 == nf  =>  nf*(1+g) + pn*(1-g)
      float o = nv * (1.f + g) + pv * (1.f - g);
      size_t oi = (size_t)r * 64 + col;
      nf_out[oi]    = o;
      nf_out_bf[oi] = f2bf(o);
    }
  }
}

// (out[src]-out[dst]) @ pe_w  ==  out[src]@pe_w + out[dst]@(-pe_w)
__global__ __launch_bounds__(256) void gemm_edge_kernel(
    const uint16_t* __restrict__ out_bf, const int* __restrict__ ei,
    const uint16_t* __restrict__ wpp, const uint16_t* __restrict__ wpn,
    const float* __restrict__ bias, float* __restrict__ ef) {
  int wid = blockIdx.x * 8 + (threadIdx.x >> 5);
  if (wid >= 20000) return;
  int lane = threadIdx.x & 31;
  int m0 = wid * 16, koff = (lane >> 4) << 3;
  int e = m0 + (lane & 15);
  const uint16_t* ps = out_bf + (size_t)ei[e] * 256;
  const uint16_t* pd = out_bf + (size_t)dst_of(ei, e) * 256;
  v8f acc[4] = {{}, {}, {}, {}};
#pragma unroll
  for (int kt = 0; kt < 8; ++kt) {
    int kb = kt * 32 + koff;
    v16bf as = load_a_frag(ps, kb);
    v16bf ad = load_a_frag(pd, kb);
#pragma unroll
    for (int t = 0; t < 4; ++t) {
      size_t ti = (((size_t)kt * 4 + t) * 32 + lane) * 16;
      v16bf bp = *(const v16bf*)(wpp + ti);
      v16bf bn = *(const v16bf*)(wpn + ti);
      acc[t] = __builtin_amdgcn_wmma_f32_16x16x32_bf16(false, as, false, bp, (short)0,
                                                       acc[t], false, false);
      acc[t] = __builtin_amdgcn_wmma_f32_16x16x32_bf16(false, ad, false, bn, (short)0,
                                                       acc[t], false, false);
    }
  }
  int rbase = m0 + koff;
#pragma unroll
  for (int t = 0; t < 4; ++t) {
    int col = t * 16 + (lane & 15);
    float b = bias[col];
#pragma unroll
    for (int j = 0; j < 8; ++j) {
      size_t o = (size_t)(rbase + j) * 64 + col;
      ef[o] += acc[t][j] + b;       // fused +e0 residual, in-place (element owned by lane)
    }
  }
}

__global__ __launch_bounds__(256) void gemm_clf_kernel(
    const uint16_t* __restrict__ nf_bf, const int* __restrict__ ei,
    const uint16_t* __restrict__ wp, const float* __restrict__ b1,
    const float* __restrict__ w2, float* __restrict__ outp) {
  int wid = blockIdx.x * 8 + (threadIdx.x >> 5);
  if (wid >= 10000) return;
  int lane = threadIdx.x & 31;
  int m0 = wid * 16, koff = (lane >> 4) << 3;
  int e = m0 + (lane & 15);
  const uint16_t* p0 = nf_bf + (size_t)ei[e] * 64;
  const uint16_t* p1 = nf_bf + (size_t)ei[e + N_EDGES] * 64;
  v8f acc[4] = {{}, {}, {}, {}};
#pragma unroll
  for (int kt = 0; kt < 4; ++kt) {
    v16bf af = (kt < 2) ? load_a_frag(p0, kt * 32 + koff)
                        : load_a_frag(p1, (kt - 2) * 32 + koff);
#pragma unroll
    for (int t = 0; t < 4; ++t) {
      v16bf bfrag = *(const v16bf*)(wp + (((size_t)kt * 4 + t) * 32 + lane) * 16);
      acc[t] = __builtin_amdgcn_wmma_f32_16x16x32_bf16(false, af, false, bfrag, (short)0,
                                                       acc[t], false, false);
    }
  }
  int rbase = m0 + koff;
#pragma unroll
  for (int t = 0; t < 4; ++t) {
    int col = t * 16 + (lane & 15);
    float cb = b1[col];
    float cw = w2[col];
#pragma unroll
    for (int j = 0; j < 8; ++j) {
      float h = fmaxf(acc[t][j] + cb, 0.f);   // relu(h)
      float p = h * cw;                        // fused h @ c_w2 partial
#pragma unroll
      for (int m = 1; m < 16; m <<= 1) p += __shfl_xor(p, m, 32);  // half-wave reduce
      if ((lane & 15) == 0) atomicAdd(&outp[rbase + j], p);
    }
  }
}

// ---------------------------------------------------------------------------
extern "C" void kernel_launch(void* const* d_in, const int* in_sizes, int n_in,
                              void* d_out, int out_size, void* d_ws, size_t ws_size,
                              hipStream_t stream) {
  (void)in_sizes; (void)n_in; (void)out_size; (void)ws_size;
  const float* x         = (const float*)d_in[0];
  const float* edge_attr = (const float*)d_in[1];
  const float* node_w    = (const float*)d_in[2];
  const float* node_b    = (const float*)d_in[3];
  const float* edge_w    = (const float*)d_in[4];
  const float* edge_b    = (const float*)d_in[5];
  const float* z_emb     = (const float*)d_in[6];
  const float* cn_w1 = (const float*)d_in[7],  *cn_b1 = (const float*)d_in[8];
  const float* cn_w2 = (const float*)d_in[9],  *cn_b2 = (const float*)d_in[10];
  const float* cn_cw = (const float*)d_in[11], *cn_cb = (const float*)d_in[12];
  const float* ce_w1 = (const float*)d_in[13], *ce_b1 = (const float*)d_in[14];
  const float* ce_w2 = (const float*)d_in[15], *ce_b2 = (const float*)d_in[16];
  const float* ce_cw = (const float*)d_in[17], *ce_cb = (const float*)d_in[18];
  const float* Wq = (const float*)d_in[19], *bq = (const float*)d_in[20];
  const float* Wk = (const float*)d_in[21], *bk = (const float*)d_in[22];
  const float* Wv = (const float*)d_in[23], *bv = (const float*)d_in[24];
  const float* Ws = (const float*)d_in[25], *bs = (const float*)d_in[26];
  const float* lt_g = (const float*)d_in[27], *lt_b = (const float*)d_in[28];
  const float* lc_g = (const float*)d_in[29], *lc_b = (const float*)d_in[30];
  const float* pe_w = (const float*)d_in[31], *pe_b = (const float*)d_in[32];
  const float* pn_w = (const float*)d_in[33], *pn_b = (const float*)d_in[34];
  const float* gf_w = (const float*)d_in[35], *gf_b = (const float*)d_in[36];
  const float* c_w1 = (const float*)d_in[37], *c_b1 = (const float*)d_in[38];
  const float* c_w2 = (const float*)d_in[39], *c_b2 = (const float*)d_in[40];
  const int*   ei   = (const int*)d_in[41];
  float* dout = (float*)d_out;

  // ---- bump allocator over d_ws ----
  char* wsb = (char*)d_ws;
  size_t off = 0;
  auto alloc_b = [&](size_t bytes) -> void* {
    void* p = wsb + off;
    off += (bytes + 255) & ~(size_t)255;
    return p;
  };
  uint16_t* pk_qkvs   = (uint16_t*)alloc_b((size_t)3 * 4 * 128 * 256 * 2);  // [iter][mat]
  uint16_t* pk_pn     = (uint16_t*)alloc_b((size_t)3 * 256 * 64 * 2);
  uint16_t* pk_pe_pos = (uint16_t*)alloc_b((size_t)3 * 256 * 64 * 2);
  uint16_t* pk_pe_neg = (uint16_t*)alloc_b((size_t)3 * 256 * 64 * 2);
  uint16_t* pk_gf     = (uint16_t*)alloc_b((size_t)3 * 128 * 64 * 2);
  uint16_t* pk_cw1    = (uint16_t*)alloc_b((size_t)128 * 64 * 2);
  float* nfA  = (float*)alloc_b((size_t)N_NODES * 64 * 4);
  float* nfB  = (float*)alloc_b((size_t)N_NODES * 64 * 4);
  uint16_t* nfA_bf = (uint16_t*)alloc_b((size_t)N_NODES * 64 * 2);
  uint16_t* nfB_bf = (uint16_t*)alloc_b((size_t)N_NODES * 64 * 2);
  float* ef   = (float*)alloc_b((size_t)N_E2 * 64 * 4);
  float* agg  = (float*)alloc_b((size_t)N_NODES * 64 * 4);
  uint16_t* comb_bf = (uint16_t*)alloc_b((size_t)N_NODES * 128 * 2);
  float* qb   = (float*)alloc_b((size_t)N_NODES * 256 * 4);
  float* kb   = (float*)alloc_b((size_t)N_NODES * 256 * 4);
  float* vb   = (float*)alloc_b((size_t)N_NODES * 256 * 4);
  float* outb = (float*)alloc_b((size_t)N_NODES * 256 * 4);   // s -> attn accumulation
  uint16_t* out_bf = (uint16_t*)alloc_b((size_t)N_NODES * 256 * 2);  // LN(out), bf16
  float* pnb  = (float*)alloc_b((size_t)N_NODES * 64 * 4);
  uint16_t* pn_bf = (uint16_t*)alloc_b((size_t)N_NODES * 64 * 2);
  float* alp  = (float*)alloc_b((size_t)N_E2 * 4 * 4);
  float* amax = (float*)alloc_b((size_t)N_NODES * 4 * 4);
  float* asum = (float*)alloc_b((size_t)N_NODES * 4 * 4);
  float* partial  = (float*)alloc_b((size_t)640 * 128 * 4);
  float* chv      = (float*)alloc_b(64 * 4);
  float* rowstats = (float*)alloc_b((size_t)N_E2 * 2 * 4);

  // ---- pre-pack weights to WMMA bf16 B-fragment layout ----
  const float* qkvsW[4] = {Wq, Wk, Wv, Ws};
  for (int it = 0; it < 3; ++it) {
    for (int m = 0; m < 4; ++m)
      pack_w_kernel<<<(128 * 256 + 255) / 256, 256, 0, stream>>>(
          qkvsW[m] + (size_t)it * 128 * 256,
          pk_qkvs + ((size_t)it * 4 + m) * 128 * 256, 128, 256, 1.f);
    pack_w_kernel<<<(256 * 64 + 255) / 256, 256, 0, stream>>>(
        pn_w + (size_t)it * 256 * 64, pk_pn + (size_t)it * 256 * 64, 256, 64, 1.f);
    pack_w_kernel<<<(256 * 64 + 255) / 256, 256, 0, stream>>>(
        pe_w + (size_t)it * 256 * 64, pk_pe_pos + (size_t)it * 256 * 64, 256, 64, 1.f);
    pack_w_kernel<<<(256 * 64 + 255) / 256, 256, 0, stream>>>(
        pe_w + (size_t)it * 256 * 64, pk_pe_neg + (size_t)it * 256 * 64, 256, 64, -1.f);
    pack_w_kernel<<<(128 * 64 + 255) / 256, 256, 0, stream>>>(
        gf_w + (size_t)it * 128 * 64, pk_gf + (size_t)it * 128 * 64, 128, 64, 1.f);
  }
  pack_w_kernel<<<(128 * 64 + 255) / 256, 256, 0, stream>>>(c_w1, pk_cw1, 128, 64, 1.f);

  // ---- embeddings + CBAM ----
  node_embed_kernel<<<(N_NODES * 64 + 255) / 256, 256, 0, stream>>>(x, node_w, node_b, z_emb, nfA);
  edge_embed_kernel<<<(N_E2 * 64 + 255) / 256, 256, 0, stream>>>(x, edge_attr, ei, edge_w,
                                                                 edge_b, ef);
  auto run_cbam = [&](float* X, int nrows, const float* w1, const float* b1, const float* w2,
                      const float* b2, const float* cw, const float* cb) {
    int nblk = (nrows + 511) / 512;
    cbam_colreduce_kernel<<<nblk, 64, 0, stream>>>(X, partial, nrows);
    cbam_stage2_kernel<<<1, 64, 0, stream>>>(partial, nblk, w1, b1, w2, b2, nrows, chv);
    int tb = (nrows + 255) / 256;
    cbam_rowscale_kernel<<<tb, 256, 0, stream>>>(X, chv, rowstats, nrows);
    cbam_spatial_kernel<<<tb, 256, 0, stream>>>(X, rowstats, cw, cb, nrows);
  };
  run_cbam(nfA, N_NODES, cn_w1, cn_b1, cn_w2, cn_b2, cn_cw, cn_cb);
  run_cbam(ef, N_E2, ce_w1, ce_b1, ce_w2, ce_b2, ce_cw, ce_cb);
  convert_bf_kernel<<<(N_NODES * 64 + 255) / 256, 256, 0, stream>>>(nfA, nfA_bf, N_NODES * 64);

  // ---- message-passing iterations ----
  float*    nf_cur    = nfA;    float*    nf_nxt    = nfB;
  uint16_t* nf_cur_bf = nfA_bf; uint16_t* nf_nxt_bf = nfB_bf;
  for (int it = 0; it < 3; ++it) {
    fill_kernel<<<(N_NODES * 64 + 255) / 256, 256, 0, stream>>>(agg, 0.f, N_NODES * 64);
    scatter_agg_kernel<<<(N_E2 * 64 + 255) / 256, 256, 0, stream>>>(ef, ei, agg);
    comb_ln_kernel<<<(N_NODES + 7) / 8, 256, 0, stream>>>(nf_cur, agg, lc_g + it * 128,
                                                          lc_b + it * 128, comb_bf);
    gemm_qkvs_kernel<<<dim3(625, 4), 256, 0, stream>>>(
        comb_bf, pk_qkvs + (size_t)it * 4 * 128 * 256,
        bq + it * 256, bk + it * 256, bv + it * 256, bs + it * 256, qb, kb, vb, outb);
    fill_kernel<<<(N_NODES * 4 + 255) / 256, 256, 0, stream>>>(amax, -3.402823466e38f,
                                                               N_NODES * 4);
    fill_kernel<<<(N_NODES * 4 + 255) / 256, 256, 0, stream>>>(asum, 0.f, N_NODES * 4);
    attn_alpha_kernel<<<(N_E2 * 4 + 255) / 256, 256, 0, stream>>>(qb, kb, ei, alp, amax);
    attn_expsum_kernel<<<(N_E2 * 4 + 255) / 256, 256, 0, stream>>>(alp, amax, ei, asum);
    attn_scatter_kernel<<<(N_E2 * 256 + 255) / 256, 256, 0, stream>>>(alp, asum, vb, ei, outb);
    ln256_kernel<<<(N_NODES + 7) / 8, 256, 0, stream>>>(outb, lt_g + it * 256, lt_b + it * 256,
                                                        out_bf);
    gemm_pn_kernel<<<157, 256, 0, stream>>>(out_bf, pk_pn + (size_t)it * 256 * 64,
                                            pn_b + it * 64, pnb, pn_bf);
    gemm_edge_kernel<<<2500, 256, 0, stream>>>(out_bf, ei,
                                               pk_pe_pos + (size_t)it * 256 * 64,
                                               pk_pe_neg + (size_t)it * 256 * 64,
                                               pe_b + it * 64, ef);
    gemm_gate_kernel<<<157, 256, 0, stream>>>(nf_cur, pnb, nf_cur_bf, pn_bf,
                                              pk_gf + (size_t)it * 128 * 64,
                                              gf_b + it * 64, nf_nxt, nf_nxt_bf);
    float* t = nf_cur; nf_cur = nf_nxt; nf_nxt = t;
    uint16_t* tb2 = nf_cur_bf; nf_cur_bf = nf_nxt_bf; nf_nxt_bf = tb2;
  }

  // ---- classifier (fused [128]@[128,64] relu @[64,1] + bias) ----
  fill_from_ptr_kernel<<<(N_EDGES + 255) / 256, 256, 0, stream>>>(dout, c_b2, N_EDGES);
  gemm_clf_kernel<<<1250, 256, 0, stream>>>(nf_cur_bf, ei, pk_cw1, c_b1, c_w2, dout);
}
